// ResBlockGnn_85469849190402
// MI455X (gfx1250) — matrix operand order repaired and checked by reading, not compiled, not run on hardware
//
#include <hip/hip_runtime.h>
#include <hip/hip_bf16.h>

#define N_NODES   100000
#define N_EDGES   1600000
#define DFEAT     128
#define NGRAPHS   512
#define NCLASSES  10
#define PADK      136   // padded K stride (bf16 elems): 272B rows -> conflict-free b128 LDS reads
#define TILES_PER_BLOCK 5
#define GEMM_BLOCKS (N_NODES / (16 * TILES_PER_BLOCK))   // 1250

typedef __attribute__((ext_vector_type(16))) __bf16        v16bf;
typedef __attribute__((ext_vector_type(8)))  float         v8f;
typedef __attribute__((ext_vector_type(4)))  unsigned int  v4u;

union Frag { v16bf bf; v4u u[2]; };

__device__ __forceinline__ __bf16 f2bf(float x) {
  union { float f; unsigned u; } v; v.f = x;
  unsigned r = v.u + 0x7FFFu + ((v.u >> 16) & 1u);   // round-to-nearest-even
  union { unsigned short s; __bf16 b; } o; o.s = (unsigned short)(r >> 16);
  return o.b;
}

// ---------------- degree / normalization ----------------
__global__ __launch_bounds__(256) void deg_kernel(const int* __restrict__ dst,
                                                  float* __restrict__ deg) {
  int e = blockIdx.x * 256 + threadIdx.x;
  if (e < N_EDGES) atomicAdd(&deg[dst[e]], 1.0f);
}

__global__ __launch_bounds__(256) void norm_kernel(const float* __restrict__ deg,
                                                   float* __restrict__ dis,
                                                   float* __restrict__ sc) {
  int n = blockIdx.x * 256 + threadIdx.x;
  if (n < N_NODES) {
    float d = deg[n] + 1.0f;      // +1 self-loop
    float r = rsqrtf(d);
    dis[n] = r;
    sc[n]  = r * r;
  }
}

__global__ __launch_bounds__(256) void ec_kernel(const int* __restrict__ src,
                                                 const int* __restrict__ dst,
                                                 const float* __restrict__ dis,
                                                 float* __restrict__ ec) {
  int e = blockIdx.x * 256 + threadIdx.x;
  if (e < N_EDGES) ec[e] = dis[src[e]] * dis[dst[e]];
}

// ---------------- per-layer weight prep: W f32 [k][n] -> bf16 transposed Wt[n][k] ----------------
__global__ __launch_bounds__(256) void wconv_kernel(const float* __restrict__ W,
                                                    __bf16* __restrict__ Wt) {
  int i = blockIdx.x * 256 + threadIdx.x;   // 64 blocks x 256 = 16384
  int k = i >> 7, n = i & 127;
  Wt[n * 128 + k] = f2bf(W[i]);
}

// ---------------- hw = (in1 [+ in2]) @ W   via v_wmma_f32_16x16x32_bf16 ----------------
// Block: 256 threads = 8 waves; wave w -> cols [16w,16w+16).
// sW staged once per block via GLOBAL_LOAD_ASYNC_TO_LDS_B128 (ASYNCcnt path);
// B fragments for all 4 K-steps are then held in registers across the tile loop.
// sA double-buffered -> one barrier per tile.
template <bool HAS_RES>
__global__ __launch_bounds__(256)
void gcn_gemm_wmma(const float* __restrict__ in1, const float* __restrict__ in2,
                   const __bf16* __restrict__ Wt, float* __restrict__ out) {
  __shared__ __bf16 sW[128 * PADK];      // sW[n*PADK + k]
  __shared__ __bf16 sA[2][16 * PADK];    // double-buffered A tile: sA[buf][m*PADK + k]

  const int tid = threadIdx.x;

  // Async-stage pre-converted Wt (bf16 [n][k]) into padded LDS: 8 x 16B per lane,
  // DMA'd cache->LDS without touching VGPRs; tracked by ASYNCcnt.
  #pragma unroll
  for (int i = 0; i < 8; ++i) {
    int chunk = i * 256 + tid;        // 2048 chunks of 8 bf16
    int n  = chunk >> 4;              // 16 chunks per 128-elem row
    int k8 = (chunk & 15) * 8;
    unsigned ldsoff = (unsigned)(size_t)&sW[n * PADK + k8];   // low 32b of flat = LDS addr
    unsigned long long ga = (unsigned long long)(size_t)(Wt + n * 128 + k8);
    asm volatile("global_load_async_to_lds_b128 %0, %1, off"
                 :: "v"(ldsoff), "v"(ga) : "memory");
  }

  const int lane = tid & 31;
  const int wave = tid >> 5;
  const int col0 = wave * 16;
  const int sel  = lane >> 4;   // K-half select per ISA A/B layouts
  const int half = lane & 15;   // M row (A) / N col (B)
  const int tile0 = blockIdx.x * TILES_PER_BLOCK;

  // Publish sW, then preload all 4 B fragments into registers (invariant per block).
  asm volatile("s_wait_asynccnt 0x0" ::: "memory");
  __syncthreads();
  Frag b[4];
  #pragma unroll
  for (int j = 0; j < 4; ++j) {
    // B 32x16 bf16: lane = column n, holds K=32j+16*sel .. +15 contiguous
    const __bf16* bp = &sW[(col0 + half) * PADK + j * 32 + sel * 16];
    b[j].u[0] = *(const v4u*)(bp);
    b[j].u[1] = *(const v4u*)(bp + 8);
  }

  for (int t = 0; t < TILES_PER_BLOCK; ++t) {
    const int row0 = (tile0 + t) * 16;
    __bf16* sa = sA[t & 1];

    // Stage A tile (16 x 128) with fused residual add, f32 -> bf16 (branch-free).
    #pragma unroll
    for (int i = 0; i < 8; ++i) {
      int idx = i * 256 + tid;
      int m = idx >> 7, k = idx & 127;
      size_t g = (size_t)(row0 + m) * DFEAT + k;
      float v = in1[g];
      if constexpr (HAS_RES) v += in2[g];
      sa[m * PADK + k] = f2bf(v);
    }
    // Prefetch next tile's A rows (global_prefetch_b8) to overlap with WMMA.
    if (t + 1 < TILES_PER_BLOCK) {
      size_t gn = (size_t)(row0 + 16) * DFEAT + tid * 8;   // 256 thr x 8 f32 = 2048
      __builtin_prefetch(in1 + gn, 0, 0);
      if constexpr (HAS_RES) __builtin_prefetch(in2 + gn, 0, 0);
    }
    __syncthreads();   // publish sa; prior-tile readers finished before their own stage

    v8f acc = {};
    #pragma unroll
    for (int j = 0; j < 4; ++j) {
      Frag a;
      // A 16x32 bf16: lane = row m, holds K=32j+16*sel .. +15 contiguous
      const __bf16* ap = &sa[half * PADK + j * 32 + sel * 16];
      a.u[0] = *(const v4u*)(ap);
      a.u[1] = *(const v4u*)(ap + 8);
      acc = __builtin_amdgcn_wmma_f32_16x16x32_bf16(false, a.bf, false, b[j].bf,
                                                    (short)0, acc, false, false);
    }

    // C layout: VGPR r -> M = r + 8*sel, N = half
    float* o = out + (size_t)(row0 + sel * 8) * DFEAT + col0 + half;
    #pragma unroll
    for (int r = 0; r < 8; ++r) o[(size_t)r * DFEAT] = acc[r];
  }
}

// ---------------- edge scatter-gather: agg[dst] += hw[src] * ec ----------------
// 1 wave per edge; lane handles 4 consecutive features (float4 gather, 4 L2 atomics).
__global__ __launch_bounds__(256)
void edge_aggregate(const int* __restrict__ src, const int* __restrict__ dst,
                    const float* __restrict__ ec, const float* __restrict__ hw,
                    float* __restrict__ agg) {
  const int  lane = threadIdx.x & 31;
  const long e    = (long)blockIdx.x * 8 + (threadIdx.x >> 5);
  if (e >= N_EDGES) return;
  const int   s = src[e], d = dst[e];
  const float c = ec[e];
  const float4 h = *(const float4*)(hw + (size_t)s * DFEAT + lane * 4);
  float* o = agg + (size_t)d * DFEAT + lane * 4;
  atomicAdd(o + 0, h.x * c);
  atomicAdd(o + 1, h.y * c);
  atomicAdd(o + 2, h.z * c);
  atomicAdd(o + 3, h.w * c);
}

// ---------------- out = relu(agg + sc*hw + b) ----------------
__global__ __launch_bounds__(256)
void epilogue_kernel(const float* __restrict__ agg, const float* __restrict__ hw,
                     const float* __restrict__ sc, const float* __restrict__ bias,
                     float* __restrict__ out) {
  int i = blockIdx.x * 256 + threadIdx.x;     // grid sized exactly N_NODES*128/256
  int node = i >> 7, f = i & 127;
  float v = agg[i] + sc[node] * hw[i] + bias[f];
  out[i] = v > 0.f ? v : 0.f;
}

// ---------------- global mean pool (atomic) ----------------
__global__ __launch_bounds__(256)
void pool_kernel(const float* __restrict__ x, const int* __restrict__ batch,
                 float* __restrict__ sums, float* __restrict__ cnts) {
  int i = blockIdx.x * 256 + threadIdx.x;     // exact N_NODES*128 threads
  int node = i >> 7, f = i & 127;
  int g = batch[node];
  atomicAdd(&sums[g * DFEAT + f], x[i]);
  if (f == 0) atomicAdd(&cnts[g], 1.0f);
}

// ---------------- head: out = (sums/cnt) @ linW + linb ----------------
__global__ __launch_bounds__(256)
void head_kernel(const float* __restrict__ sums, const float* __restrict__ cnts,
                 const float* __restrict__ linW, const float* __restrict__ linb,
                 float* __restrict__ out) {
  int i = blockIdx.x * 256 + threadIdx.x;
  if (i >= NGRAPHS * NCLASSES) return;
  int g = i / NCLASSES, c = i % NCLASSES;
  float inv = 1.0f / fmaxf(cnts[g], 1.0f);
  float acc = linb[c];
  const float* s = sums + g * DFEAT;
  #pragma unroll 8
  for (int f = 0; f < DFEAT; ++f) acc += (s[f] * inv) * linW[f * NCLASSES + c];
  out[i] = acc;
}

// ---------------------------------------------------------------------------
extern "C" void kernel_launch(void* const* d_in, const int* in_sizes, int n_in,
                              void* d_out, int out_size, void* d_ws, size_t ws_size,
                              hipStream_t stream) {
  (void)in_sizes; (void)n_in; (void)out_size; (void)ws_size;

  const float* x     = (const float*)d_in[0];
  const int*   eidx  = (const int*)d_in[1];
  const int*   src   = eidx;
  const int*   dst   = eidx + N_EDGES;
  const int*   batch = (const int*)d_in[2];
  const float* W0    = (const float*)d_in[3];
  const float* b0    = (const float*)d_in[4];
  const float* Ws    = (const float*)d_in[5];   // [3][128][128]
  const float* bs    = (const float*)d_in[6];   // [3][128]
  const float* linW  = (const float*)d_in[7];
  const float* linb  = (const float*)d_in[8];
  float*       outp  = (float*)d_out;

  // workspace layout
  char* ws = (char*)d_ws;
  size_t off = 0;
  auto take = [&](size_t bytes) -> void* {
    void* p = ws + off;
    off += (bytes + 255) & ~(size_t)255;
    return p;
  };
  const size_t NF = (size_t)N_NODES * DFEAT * sizeof(float);
  float*  deg  = (float*)take((size_t)N_NODES * 4);
  float*  dis  = (float*)take((size_t)N_NODES * 4);
  float*  sc   = (float*)take((size_t)N_NODES * 4);
  float*  ec   = (float*)take((size_t)N_EDGES * 4);
  float*  hw   = (float*)take(NF);
  float*  agg  = (float*)take(NF);
  float*  B0   = (float*)take(NF);
  float*  B1   = (float*)take(NF);
  __bf16* Wt   = (__bf16*)take((size_t)128 * 128 * 2);
  float*  sums = (float*)take((size_t)NGRAPHS * DFEAT * 4);
  float*  cnts = (float*)take((size_t)NGRAPHS * 4);

  // normalization coefficients
  hipMemsetAsync(deg, 0, (size_t)N_NODES * 4, stream);
  deg_kernel <<<N_EDGES / 256, 256, 0, stream>>>(dst, deg);
  norm_kernel<<<(N_NODES + 255) / 256, 256, 0, stream>>>(deg, dis, sc);
  ec_kernel  <<<N_EDGES / 256, 256, 0, stream>>>(src, dst, dis, ec);

  auto run_layer = [&](const float* inA, const float* inB,
                       const float* Wp, const float* bp, float* outBuf) {
    wconv_kernel<<<64, 256, 0, stream>>>(Wp, Wt);
    if (inB)
      gcn_gemm_wmma<true> <<<GEMM_BLOCKS, 256, 0, stream>>>(inA, inB, Wt, hw);
    else
      gcn_gemm_wmma<false><<<GEMM_BLOCKS, 256, 0, stream>>>(inA, nullptr, Wt, hw);
    hipMemsetAsync(agg, 0, NF, stream);
    edge_aggregate <<<N_EDGES / 8, 256, 0, stream>>>(src, dst, ec, hw, agg);
    epilogue_kernel<<<(N_NODES * DFEAT) / 256, 256, 0, stream>>>(agg, hw, sc, bp, outBuf);
  };

  // layer 0: C0 = relu(conv(x, W0))            -> B0
  run_layer(x,  nullptr, W0,            b0,       B0);
  // i=0: C1 = relu(conv(C0 + 0, Ws[0]))        -> B1   (x_pre was zeros)
  run_layer(B0, nullptr, Ws,            bs,       B1);
  // i=1: C2 = relu(conv(C1 + C0, Ws[1]))       -> B0   (B0 dead after GEMM read)
  run_layer(B1, B0,      Ws + 16384,    bs + 128, B0);
  // i=2: C3 = relu(conv(C2 + C1, Ws[2]))       -> B1
  run_layer(B0, B1,      Ws + 32768,    bs + 256, B1);

  // mean pool + linear head
  hipMemsetAsync(sums, 0, (size_t)NGRAPHS * DFEAT * 4, stream);
  hipMemsetAsync(cnts, 0, (size_t)NGRAPHS * 4, stream);
  pool_kernel<<<(N_NODES * DFEAT) / 256, 256, 0, stream>>>(B1, batch, sums, cnts);
  head_kernel<<<(NGRAPHS * NCLASSES + 255) / 256, 256, 0, stream>>>(sums, cnts, linW, linb, outp);
}
